// MeshGraphNet_35390530519053
// MI455X (gfx1250) — matrix operand order repaired and checked by reading, not compiled, not run on hardware
//
#include <hip/hip_runtime.h>
#include <hip/hip_bf16.h>

// ---------------------------------------------------------------------------
// MeshGraphNet forward for MI455X (gfx1250).
//  - All 128-wide GEMMs via v_wmma_f32_16x16x32_bf16 (fp32 accum).
//  - segment_sum via once-per-call CSR build + gather/sum (no per-layer atomics).
//  - Double-buffered LDS A-tiles; B-matrices pre-packed in per-lane WMMA
//    fragment layout so every lane fetches 32B contiguous (L2-resident).
//  - LayerNorm fully in registers via 16-lane shfl_xor reductions.
// ---------------------------------------------------------------------------

typedef __attribute__((ext_vector_type(16))) __bf16 v16bf;
typedef __attribute__((ext_vector_type(8)))  float  v8f;
typedef unsigned short u16;

union Frag {
    uint4 u[2];
    v16bf v;
};

#define WMMA_BF16(a, b, c) \
    __builtin_amdgcn_wmma_f32_16x16x32_bf16(false, (a), false, (b), (short)0, (c), false, false)

__device__ __forceinline__ u16 f2bf(float f) {
    unsigned u = __float_as_uint(f);
    u += 0x7fffu + ((u >> 16) & 1u);          // round-to-nearest-even
    return (u16)(u >> 16);
}

__device__ __forceinline__ float bf2f(u16 b) {
    return __uint_as_float(((unsigned)b) << 16);
}

__device__ __forceinline__ float silu_f(float x) { return x / (1.0f + __expf(-x)); }

__device__ __forceinline__ float redsum16(float x) {
    x += __shfl_xor(x, 1, 16);
    x += __shfl_xor(x, 2, 16);
    x += __shfl_xor(x, 4, 16);
    x += __shfl_xor(x, 8, 16);
    return x;
}

// A-fragment (16x32 bf16, MxK) from LDS, row-major with element stride strideE.
// lanes 0-15: M=lane, K {0..7,16..23}; lanes 16-31: M=lane-16, K {8..15,24..31}
__device__ __forceinline__ v16bf load_a_frag(const u16* base, int strideE, int m0, int kbase) {
    int lane = threadIdx.x & 31;
    const u16* p = base + (m0 + (lane & 15)) * strideE + kbase + ((lane >> 4) << 3);
    Frag f;
    f.u[0] = *(const uint4*)(p);        // K+0..7   (or 8..15)
    f.u[1] = *(const uint4*)(p + 16);   // K+16..23 (or 24..31)
    return f.v;
}

// B-fragment from pre-packed global buffer: each lane's 16 values contiguous (32B).
__device__ __forceinline__ v16bf load_b_frag(const u16* pk, int NT, int kc, int nt) {
    int lane = threadIdx.x & 31;
    const u16* p = pk + ((size_t)((kc * NT + nt) * 32 + lane) << 4);
    Frag f;
    f.u[0] = *(const uint4*)(p);
    f.u[1] = *(const uint4*)(p + 8);
    return f.v;
}

// ---------------------------------------------------------------------------
// Pack fp32 weight matrix W[K][N] (row-major) into bf16 WMMA B-fragments.
// ---------------------------------------------------------------------------
__global__ void pack_b_kernel(const float* __restrict__ W, u16* __restrict__ dst, int K, int N) {
    int t = blockIdx.x * blockDim.x + threadIdx.x;
    int NT = N >> 4, KC = K >> 5;
    if (t >= KC * NT * 32) return;
    int lane = t & 31;
    int tile = t >> 5;
    int nt = tile % NT;
    int kc = tile / NT;
    int n  = (nt << 4) + (lane & 15);
    int k0 = (kc << 5) + ((lane >> 4) << 3);
    u16* o = dst + ((size_t)t << 4);
#pragma unroll
    for (int j = 0; j < 16; ++j) o[j] = f2bf(W[(size_t)(k0 + j + ((j >> 3) << 3)) * N + n]);
}

// ---------------------------------------------------------------------------
// CSR build: degree histogram -> 3-phase exclusive scan -> atomic fill.
// ---------------------------------------------------------------------------
__global__ void deg_kernel(const int* __restrict__ dstI, int* __restrict__ deg, int E) {
    int e = blockIdx.x * blockDim.x + threadIdx.x;
    if (e < E) atomicAdd(&deg[dstI[e]], 1);
}

#define SCAN_B 1024   // elements per scan block (256 threads x 4)

__global__ __launch_bounds__(256) void scan_phase1(const int* __restrict__ deg,
                                                   int* __restrict__ bsum, int n) {
    __shared__ int sh[256];
    int b0 = blockIdx.x * SCAN_B;
    int local = 0;
#pragma unroll
    for (int j = 0; j < 4; ++j) {
        int i = b0 + threadIdx.x * 4 + j;
        local += (i < n) ? deg[i] : 0;
    }
    sh[threadIdx.x] = local;
    __syncthreads();
    for (int off = 128; off > 0; off >>= 1) {
        if (threadIdx.x < (unsigned)off) sh[threadIdx.x] += sh[threadIdx.x + off];
        __syncthreads();
    }
    if (threadIdx.x == 0) bsum[blockIdx.x] = sh[0];
}

__global__ void scan_phase2(int* __restrict__ bsum, int nb, int* __restrict__ rowptr_last) {
    if (threadIdx.x == 0 && blockIdx.x == 0) {
        int run = 0;
        for (int i = 0; i < nb; ++i) { int v = bsum[i]; bsum[i] = run; run += v; }
        *rowptr_last = run;
    }
}

__global__ __launch_bounds__(256) void scan_phase3(const int* __restrict__ deg,
                                                   const int* __restrict__ bsum,
                                                   int* __restrict__ rowptr, int n) {
    __shared__ int sh[256];
    int b0 = blockIdx.x * SCAN_B;
    int v[4];
    int local = 0;
#pragma unroll
    for (int j = 0; j < 4; ++j) {
        int i = b0 + threadIdx.x * 4 + j;
        v[j] = (i < n) ? deg[i] : 0;
        local += v[j];
    }
    sh[threadIdx.x] = local;
    __syncthreads();
    for (int off = 1; off < 256; off <<= 1) {
        int x = (threadIdx.x >= (unsigned)off) ? sh[threadIdx.x - off] : 0;
        __syncthreads();
        sh[threadIdx.x] += x;
        __syncthreads();
    }
    int excl = sh[threadIdx.x] - local + bsum[blockIdx.x];
#pragma unroll
    for (int j = 0; j < 4; ++j) {
        int i = b0 + threadIdx.x * 4 + j;
        if (i < n) rowptr[i] = excl;
        excl += v[j];
    }
}

__global__ void csr_fill_kernel(const int* __restrict__ dstI, int* __restrict__ cursor,
                                int* __restrict__ eid, int E) {
    int e = blockIdx.x * blockDim.x + threadIdx.x;
    if (e >= E) return;
    int p = atomicAdd(&cursor[dstI[e]], 1);
    eid[p] = e;
}

// ---------------------------------------------------------------------------
// Encoder: hidden = silu(X@w1+b1) (scalar, in_dim<=6), out = LN(hidden@w2+b2)
// ---------------------------------------------------------------------------
__global__ __launch_bounds__(256) void encoder_kernel(
    const float* __restrict__ X, int in_dim,
    const float* __restrict__ w1, const float* __restrict__ b1,
    const u16* __restrict__ w2pk, const float* __restrict__ b2,
    const float* __restrict__ g, const float* __restrict__ beta,
    float* __restrict__ out_f32, u16* __restrict__ out_bf, int Nrows)
{
    __shared__ u16   Hs[128 * 128];
    __shared__ float xs[128 * 8];
    __shared__ float w1s[8 * 128];
    __shared__ float b1s[128];

    const int t  = threadIdx.x;
    const int n0 = blockIdx.x * 128;
    const int lane = t & 31;
    const int m0 = (t >> 5) << 4;

    for (int i = t; i < in_dim * 128; i += 256) w1s[i] = w1[i];
    if (t < 128) b1s[t] = b1[t];
    for (int i = t; i < 128 * in_dim; i += 256) {
        int r = i / in_dim, c = i - r * in_dim;
        int row = n0 + r; if (row >= Nrows) row = Nrows - 1;
        xs[r * 8 + c] = X[(size_t)row * in_dim + c];
    }
    __syncthreads();

    {
        int r  = t >> 1;
        int nb = (t & 1) << 6;
        for (int n = nb; n < nb + 64; ++n) {
            float a = b1s[n];
            for (int k = 0; k < in_dim; ++k) a += xs[r * 8 + k] * w1s[k * 128 + n];
            Hs[r * 128 + n] = f2bf(silu_f(a));
        }
    }
    __syncthreads();

    v8f acc[8] = {};
#pragma unroll
    for (int kc = 0; kc < 4; ++kc) {
        v16bf a = load_a_frag(Hs, 128, m0, kc * 32);
#pragma unroll
        for (int nt = 0; nt < 8; ++nt)
            acc[nt] = WMMA_BF16(a, load_b_frag(w2pk, 8, kc, nt), acc[nt]);
    }

    float bb[8], gg[8], be[8];
#pragma unroll
    for (int nt = 0; nt < 8; ++nt) {
        int n = nt * 16 + (lane & 15);
        bb[nt] = b2[n]; gg[nt] = g[n]; be[nt] = beta[n];
    }
#pragma unroll
    for (int v = 0; v < 8; ++v) {
        float vals[8], s = 0.f, s2 = 0.f;
#pragma unroll
        for (int nt = 0; nt < 8; ++nt) {
            float x = acc[nt][v] + bb[nt];
            vals[nt] = x; s += x; s2 += x * x;
        }
        s = redsum16(s); s2 = redsum16(s2);
        float mean = s * (1.f / 128.f);
        float rn = rsqrtf(s2 * (1.f / 128.f) - mean * mean + 1e-5f);
        int m = m0 + v + ((lane >> 4) << 3);
        int gr = n0 + m;
        if (gr < Nrows) {
#pragma unroll
            for (int nt = 0; nt < 8; ++nt) {
                int n = nt * 16 + (lane & 15);
                float o = gg[nt] * (vals[nt] - mean) * rn + be[nt];
                if (out_f32) out_f32[(size_t)gr * 128 + n] = o;
                out_bf[(size_t)gr * 128 + n] = f2bf(o);
            }
        }
    }
}

// ---------------------------------------------------------------------------
// Edge message: msg = LN(W2 @ silu(W1 @ [h[dst]|h[src]|e] + b1) + b2)
// written as coalesced bf16 rows into msgbuf.  128 edges/block.
// Double-buffered gathered A-staging.
// ---------------------------------------------------------------------------
__global__ __launch_bounds__(256) void edge_msg_kernel(
    const u16* __restrict__ h_bf, const u16* __restrict__ e_bf,
    const int* __restrict__ srcI, const int* __restrict__ dstI,
    const u16* __restrict__ w1pk, const float* __restrict__ b1,
    const u16* __restrict__ w2pk, const float* __restrict__ b2,
    const float* __restrict__ g,  const float* __restrict__ beta,
    u16* __restrict__ msgbuf)
{
    __shared__ u16 As[2][128 * 32];
    __shared__ u16 Hs[128 * 128];
    __shared__ int sDst[128], sSrc[128];

    const int t  = threadIdx.x;
    const int e0 = blockIdx.x * 128;
    if (t < 128) { sDst[t] = dstI[e0 + t]; sSrc[t] = srcI[e0 + t]; }

    const int lane = t & 31;
    const int m0 = (t >> 5) << 4;
    const int r = t >> 1, halfo = (t & 1) << 4;

    __syncthreads();   // indices visible

    // stage chunk 0
    {
        const u16* sp = h_bf + (size_t)sDst[r] * 128;
        *(uint4*)(&As[0][r * 32 + halfo]) = *(const uint4*)(sp + halfo);
    }

    v8f acc[8] = {};
    for (int kc = 0; kc < 12; ++kc) {
        __syncthreads();                       // chunk kc staged & prev reads done
        if (kc < 11) {                         // prefetch chunk kc+1 into other buffer
            int kn = kc + 1;
            const u16* sp;
            if (kn < 4)      sp = h_bf + (size_t)sDst[r] * 128 + kn * 32;
            else if (kn < 8) sp = h_bf + (size_t)sSrc[r] * 128 + (kn - 4) * 32;
            else             sp = e_bf + (size_t)(e0 + r) * 128 + (kn - 8) * 32;
            *(uint4*)(&As[kn & 1][r * 32 + halfo]) = *(const uint4*)(sp + halfo);
        }
        v16bf a = load_a_frag(&As[kc & 1][0], 32, m0, 0);
#pragma unroll
        for (int nt = 0; nt < 8; ++nt)
            acc[nt] = WMMA_BF16(a, load_b_frag(w1pk, 8, kc, nt), acc[nt]);
    }

    // bias + SiLU -> bf16 hidden tile
#pragma unroll
    for (int nt = 0; nt < 8; ++nt) {
        int n = nt * 16 + (lane & 15);
        float bb = b1[n];
#pragma unroll
        for (int v = 0; v < 8; ++v) {
            int m = m0 + v + ((lane >> 4) << 3);
            Hs[m * 128 + n] = f2bf(silu_f(acc[nt][v] + bb));
        }
    }
    __syncthreads();

    v8f acc2[8] = {};
#pragma unroll
    for (int kc = 0; kc < 4; ++kc) {
        v16bf a = load_a_frag(Hs, 128, m0, kc * 32);
#pragma unroll
        for (int nt = 0; nt < 8; ++nt)
            acc2[nt] = WMMA_BF16(a, load_b_frag(w2pk, 8, kc, nt), acc2[nt]);
    }

    float bb[8], gg[8], be[8];
#pragma unroll
    for (int nt = 0; nt < 8; ++nt) {
        int n = nt * 16 + (lane & 15);
        bb[nt] = b2[n]; gg[nt] = g[n]; be[nt] = beta[n];
    }
    __syncthreads();   // all GEMM2 reads of Hs done; safe to overwrite with msg
#pragma unroll
    for (int v = 0; v < 8; ++v) {
        float vals[8], s = 0.f, s2 = 0.f;
#pragma unroll
        for (int nt = 0; nt < 8; ++nt) {
            float x = acc2[nt][v] + bb[nt];
            vals[nt] = x; s += x; s2 += x * x;
        }
        s = redsum16(s); s2 = redsum16(s2);
        float mean = s * (1.f / 128.f);
        float rn = rsqrtf(s2 * (1.f / 128.f) - mean * mean + 1e-5f);
        int m = m0 + v + ((lane >> 4) << 3);
#pragma unroll
        for (int nt = 0; nt < 8; ++nt) {
            int n = nt * 16 + (lane & 15);
            Hs[m * 128 + n] = f2bf(gg[nt] * (vals[nt] - mean) * rn + be[nt]);
        }
    }
    __syncthreads();
    // coalesced store: 2 threads per row, 128B each (8x b128)
    {
        u16* op = msgbuf + (size_t)(e0 + r) * 128 + ((t & 1) << 6);
        const u16* ip = &Hs[r * 128 + ((t & 1) << 6)];
#pragma unroll
        for (int q = 0; q < 8; ++q)
            *(uint4*)(op + q * 8) = *(const uint4*)(ip + q * 8);
    }
}

// ---------------------------------------------------------------------------
// Node update: agg = CSR-gather-sum of msgs; h += LN(W2@silu(W1@[h|agg]+b1)+b2)
// agg tile lives in LDS (Hs) as bf16 -> chunks K=128..255 need no staging.
// ---------------------------------------------------------------------------
__global__ __launch_bounds__(256) void node_upd_kernel(
    float* __restrict__ h, u16* __restrict__ h_bf,
    const u16* __restrict__ msgbuf,
    const int* __restrict__ rowptr, const int* __restrict__ eid,
    const u16* __restrict__ w1pk, const float* __restrict__ b1,
    const u16* __restrict__ w2pk, const float* __restrict__ b2,
    const float* __restrict__ g,  const float* __restrict__ beta, int Nn)
{
    __shared__ u16 As[2][128 * 32];
    __shared__ u16 Hs[128 * 128];

    const int t  = threadIdx.x;
    const int n0 = blockIdx.x * 128;
    const int lane = t & 31;
    const int m0 = (t >> 5) << 4;
    const int r = t >> 1, halfo = (t & 1) << 4;
    const int colb = (t & 1) << 6;       // this thread covers 64 columns
    int node = n0 + r;
    const bool valid = node < Nn;
    if (!valid) node = Nn - 1;

    // --- phase A: gather + fp32 sum of incident messages, 64 cols/thread
    {
        float a[64];
#pragma unroll
        for (int j = 0; j < 64; ++j) a[j] = 0.f;
        if (valid) {
            int p0 = rowptr[node], p1 = rowptr[node + 1];
            for (int p = p0; p < p1; ++p) {
                const u16* mp = msgbuf + (size_t)eid[p] * 128 + colb;
#pragma unroll
                for (int q = 0; q < 8; ++q) {
                    uint4 u = *(const uint4*)(mp + q * 8);
                    unsigned w0 = u.x, w1 = u.y, w2 = u.z, w3 = u.w;
                    a[q*8+0] += __uint_as_float(w0 << 16);
                    a[q*8+1] += __uint_as_float(w0 & 0xffff0000u);
                    a[q*8+2] += __uint_as_float(w1 << 16);
                    a[q*8+3] += __uint_as_float(w1 & 0xffff0000u);
                    a[q*8+4] += __uint_as_float(w2 << 16);
                    a[q*8+5] += __uint_as_float(w2 & 0xffff0000u);
                    a[q*8+6] += __uint_as_float(w3 << 16);
                    a[q*8+7] += __uint_as_float(w3 & 0xffff0000u);
                }
            }
        }
#pragma unroll
        for (int j = 0; j < 64; ++j) Hs[r * 128 + colb + j] = f2bf(a[j]);
    }

    // stage h chunk 0
    *(uint4*)(&As[0][r * 32 + halfo]) =
        *(const uint4*)(h_bf + (size_t)node * 128 + halfo);

    v8f acc[8] = {};
    for (int kc = 0; kc < 8; ++kc) {
        __syncthreads();
        if (kc + 1 < 4) {
            int kn = kc + 1;
            *(uint4*)(&As[kn & 1][r * 32 + halfo]) =
                *(const uint4*)(h_bf + (size_t)node * 128 + kn * 32 + halfo);
        }
        v16bf a = (kc < 4) ? load_a_frag(&As[kc & 1][0], 32, m0, 0)
                           : load_a_frag(Hs, 128, m0, (kc - 4) * 32);
#pragma unroll
        for (int nt = 0; nt < 8; ++nt)
            acc[nt] = WMMA_BF16(a, load_b_frag(w1pk, 8, kc, nt), acc[nt]);
    }
    __syncthreads();   // all agg reads of Hs done; safe to overwrite with hidden

#pragma unroll
    for (int nt = 0; nt < 8; ++nt) {
        int n = nt * 16 + (lane & 15);
        float bb = b1[n];
#pragma unroll
        for (int v = 0; v < 8; ++v) {
            int m = m0 + v + ((lane >> 4) << 3);
            Hs[m * 128 + n] = f2bf(silu_f(acc[nt][v] + bb));
        }
    }
    __syncthreads();

    v8f acc2[8] = {};
#pragma unroll
    for (int kc = 0; kc < 4; ++kc) {
        v16bf a = load_a_frag(Hs, 128, m0, kc * 32);
#pragma unroll
        for (int nt = 0; nt < 8; ++nt)
            acc2[nt] = WMMA_BF16(a, load_b_frag(w2pk, 8, kc, nt), acc2[nt]);
    }

    float bb[8], gg[8], be[8];
#pragma unroll
    for (int nt = 0; nt < 8; ++nt) {
        int n = nt * 16 + (lane & 15);
        bb[nt] = b2[n]; gg[nt] = g[n]; be[nt] = beta[n];
    }
#pragma unroll
    for (int v = 0; v < 8; ++v) {
        float vals[8], s = 0.f, s2 = 0.f;
#pragma unroll
        for (int nt = 0; nt < 8; ++nt) {
            float x = acc2[nt][v] + bb[nt];
            vals[nt] = x; s += x; s2 += x * x;
        }
        s = redsum16(s); s2 = redsum16(s2);
        float mean = s * (1.f / 128.f);
        float rn = rsqrtf(s2 * (1.f / 128.f) - mean * mean + 1e-5f);
        int m = m0 + v + ((lane >> 4) << 3);
        int gn = n0 + m;
        if (gn < Nn) {
            float* hp = h + (size_t)gn * 128;
#pragma unroll
            for (int nt = 0; nt < 8; ++nt) {
                int n = nt * 16 + (lane & 15);
                float o = hp[n] + gg[nt] * (vals[nt] - mean) * rn + be[nt];
                hp[n] = o;
                h_bf[(size_t)gn * 128 + n] = f2bf(o);
            }
        }
    }
}

// ---------------------------------------------------------------------------
// Decoder: out = silu(h@w1+b1)@w2+b2.  4 nodes/block, 64 threads/node.
// ---------------------------------------------------------------------------
__global__ __launch_bounds__(256) void decoder_kernel(
    const float* __restrict__ h, const float* __restrict__ w1, const float* __restrict__ b1,
    const float* __restrict__ w2, const float* __restrict__ b2,
    float* __restrict__ out, int Nn)
{
    __shared__ float hid[4][64];
    const int slot = threadIdx.x >> 6;
    const int j    = threadIdx.x & 63;
    int node = blockIdx.x * 4 + slot;
    int cn = node < Nn ? node : Nn - 1;
    const float* hp = h + (size_t)cn * 128;
    float a = b1[j];
    for (int k = 0; k < 128; ++k) a += hp[k] * w1[k * 64 + j];
    hid[slot][j] = silu_f(a);
    __syncthreads();
    if (node < Nn && j < 3) {
        float o = b2[j];
        const float* hh = hid[slot];
        for (int k = 0; k < 64; ++k) o += hh[k] * w2[k * 3 + j];
        out[(size_t)node * 3 + j] = o;
    }
}

// ---------------------------------------------------------------------------
extern "C" void kernel_launch(void* const* d_in, const int* in_sizes, int n_in,
                              void* d_out, int out_size, void* d_ws, size_t ws_size,
                              hipStream_t stream)
{
    (void)in_sizes; (void)n_in; (void)out_size; (void)ws_size;
    const int NN = 50000, NE = 400000;

    // --- inputs (top-level insertion order; params pytree-flattened, sorted keys)
    const float* x         = (const float*)d_in[0];
    const int*   eidx      = (const int*)d_in[1];
    const float* edge_attr = (const float*)d_in[2];
    const float* be_b1 = (const float*)d_in[3];   // blk_edge [6,128]
    const float* be_b2 = (const float*)d_in[4];
    const float* be_bt = (const float*)d_in[5];
    const float* be_g  = (const float*)d_in[6];
    const float* be_w1 = (const float*)d_in[7];   // [6,384,128]
    const float* be_w2 = (const float*)d_in[8];   // [6,128,128]
    const float* bn_b1 = (const float*)d_in[9];   // blk_node
    const float* bn_b2 = (const float*)d_in[10];
    const float* bn_bt = (const float*)d_in[11];
    const float* bn_g  = (const float*)d_in[12];
    const float* bn_w1 = (const float*)d_in[13];  // [6,256,128]
    const float* bn_w2 = (const float*)d_in[14];  // [6,128,128]
    const float* dec_b1 = (const float*)d_in[15];
    const float* dec_b2 = (const float*)d_in[16];
    const float* dec_w1 = (const float*)d_in[17]; // [128,64]
    const float* dec_w2 = (const float*)d_in[18]; // [64,3]
    const float* ee_b1 = (const float*)d_in[19];  // edge_enc
    const float* ee_b2 = (const float*)d_in[20];
    const float* ee_bt = (const float*)d_in[21];
    const float* ee_g  = (const float*)d_in[22];
    const float* ee_w1 = (const float*)d_in[23];  // [3,128]
    const float* ee_w2 = (const float*)d_in[24];  // [128,128]
    const float* ne_b1 = (const float*)d_in[25];  // node_enc
    const float* ne_b2 = (const float*)d_in[26];
    const float* ne_bt = (const float*)d_in[27];
    const float* ne_g  = (const float*)d_in[28];
    const float* ne_w1 = (const float*)d_in[29];  // [6,128]
    const float* ne_w2 = (const float*)d_in[30];  // [128,128]

    const int* srcI = eidx;        // edge_index[0] = source j
    const int* dstI = eidx + NE;   // edge_index[1] = target i

    // --- workspace carve-out (256B aligned running cursor)
    char* ws = (char*)d_ws;
    size_t cur = 0;
    auto carve = [&](size_t bytes) {
        void* p = ws + cur;
        cur += (bytes + 255) & ~(size_t)255;
        return p;
    };
    float* h      = (float*)carve((size_t)NN * 128 * 4);    // 25.6 MB
    u16*   h_bf   = (u16*)  carve((size_t)NN * 128 * 2);    // 12.8 MB
    u16*   e_bf   = (u16*)  carve((size_t)NE * 128 * 2);    // 102.4 MB
    u16*   msgbuf = (u16*)  carve((size_t)NE * 128 * 2);    // 102.4 MB
    int*   deg    = (int*)  carve((size_t)NN * 4);
    int*   rowptr = (int*)  carve((size_t)(NN + 1) * 4);
    int*   cursor = (int*)  carve((size_t)NN * 4);
    int*   bsum   = (int*)  carve(64 * 4);
    int*   eid    = (int*)  carve((size_t)NE * 4);
    u16* pk_ne_w2 = (u16*)  carve(128 * 128 * 2);
    u16* pk_ee_w2 = (u16*)  carve(128 * 128 * 2);
    u16* ew1[6]; u16* ew2[6]; u16* nw1[6]; u16* nw2[6];
    for (int l = 0; l < 6; ++l) {
        ew1[l] = (u16*)carve(384 * 128 * 2);
        ew2[l] = (u16*)carve(128 * 128 * 2);
        nw1[l] = (u16*)carve(256 * 128 * 2);
        nw2[l] = (u16*)carve(128 * 128 * 2);
    }

    // --- CSR build (graph static -> once per call)
    const int NB_SCAN = (NN + SCAN_B - 1) / SCAN_B;
    hipMemsetAsync(deg, 0, (size_t)NN * 4, stream);
    deg_kernel<<<(NE + 255) / 256, 256, 0, stream>>>(dstI, deg, NE);
    scan_phase1<<<NB_SCAN, 256, 0, stream>>>(deg, bsum, NN);
    scan_phase2<<<1, 64, 0, stream>>>(bsum, NB_SCAN, rowptr + NN);
    scan_phase3<<<NB_SCAN, 256, 0, stream>>>(deg, bsum, rowptr, NN);
    hipMemcpyAsync(cursor, rowptr, (size_t)NN * 4, hipMemcpyDeviceToDevice, stream);
    csr_fill_kernel<<<(NE + 255) / 256, 256, 0, stream>>>(dstI, cursor, eid, NE);

    // --- pack weights to bf16 WMMA B-fragments
    pack_b_kernel<<<4, 256, 0, stream>>>(ne_w2, pk_ne_w2, 128, 128);
    pack_b_kernel<<<4, 256, 0, stream>>>(ee_w2, pk_ee_w2, 128, 128);
    for (int l = 0; l < 6; ++l) {
        pack_b_kernel<<<12, 256, 0, stream>>>(be_w1 + (size_t)l * 384 * 128, ew1[l], 384, 128);
        pack_b_kernel<<<4,  256, 0, stream>>>(be_w2 + (size_t)l * 128 * 128, ew2[l], 128, 128);
        pack_b_kernel<<<8,  256, 0, stream>>>(bn_w1 + (size_t)l * 256 * 128, nw1[l], 256, 128);
        pack_b_kernel<<<4,  256, 0, stream>>>(bn_w2 + (size_t)l * 128 * 128, nw2[l], 128, 128);
    }

    // --- encoders
    encoder_kernel<<<(NN + 127) / 128, 256, 0, stream>>>(
        x, 6, ne_w1, ne_b1, pk_ne_w2, ne_b2, ne_g, ne_bt, h, h_bf, NN);
    encoder_kernel<<<NE / 128, 256, 0, stream>>>(
        edge_attr, 3, ee_w1, ee_b1, pk_ee_w2, ee_b2, ee_g, ee_bt, nullptr, e_bf, NE);

    // --- 6 message-passing layers
    for (int l = 0; l < 6; ++l) {
        edge_msg_kernel<<<NE / 128, 256, 0, stream>>>(
            h_bf, e_bf, srcI, dstI,
            ew1[l], be_b1 + l * 128, ew2[l], be_b2 + l * 128,
            be_g + l * 128, be_bt + l * 128, msgbuf);
        node_upd_kernel<<<(NN + 127) / 128, 256, 0, stream>>>(
            h, h_bf, msgbuf, rowptr, eid,
            nw1[l], bn_b1 + l * 128, nw2[l], bn_b2 + l * 128,
            bn_g + l * 128, bn_bt + l * 128, NN);
    }

    // --- decoder
    decoder_kernel<<<(NN + 3) / 4, 256, 0, stream>>>(
        h, dec_w1, dec_b1, dec_w2, dec_b2, (float*)d_out, NN);
}
